// LinearAttention_4028679324115
// MI455X (gfx1250) — compile-verified, matrix operand order checked by reading, not run on hardware
//
#include <hip/hip_runtime.h>
#include <hip/hip_bf16.h>

// ---------------------------------------------------------------------------
// Taylor linear attention (Based-style) for MI455X / gfx1250.
// B=4, L=2048, D_MODEL=1024, H=16, FEAT=16, HEAD_DIM=64, CHUNK=128,
// De = 1+16+256 = 273 (padded to 288 = 18*16 for WMMA).
// bf16 end-to-end: one fp32->bf16 conversion pass, then all GEMMs read/write
// bf16 via v_wmma_f32_16x16x32_bf16 with async global->LDS double buffering.
// ---------------------------------------------------------------------------

typedef __attribute__((ext_vector_type(16))) __bf16 v16bf;
typedef __attribute__((ext_vector_type(4)))  __bf16 v4bf;
typedef __attribute__((ext_vector_type(8)))  float v8f;

#define SEQ_L   2048
#define NBATCH  4
#define NHEADS  16
#define NFEAT   16
#define HDIM    64
#define NCHUNK  16
#define CHUNKL  128
#define DE_PAD  288

// qf = [1, q*0.5, (q_a*q_b)/(4*sqrt(2))]  (d=16: d^-0.25 = 0.5)
#define C_LIN  0.5f
#define C_QUAD 0.17677669529663687f

__device__ __forceinline__ v8f wmma_bf16(v16bf a, v16bf b, v8f c) {
  return __builtin_amdgcn_wmma_f32_16x16x32_bf16(false, a, false, b, (short)0, c,
                                                 false, false);
}

// Async 16-byte global->LDS copy (CDNA5 GLOBAL_LOAD_ASYNC_TO_LDS_B128, ASYNCcnt).
__device__ __forceinline__ void async_ld16(void* lds_dst, const void* gsrc) {
  unsigned l = (unsigned)(uintptr_t)lds_dst;               // LDS aperture: low 32 bits
  unsigned long long g = (unsigned long long)(uintptr_t)gsrc;
  asm volatile("global_load_async_to_lds_b128 %0, %1, off" :: "v"(l), "v"(g) : "memory");
}
__device__ __forceinline__ void wait_async0() {
  asm volatile("s_wait_asynccnt 0x0" ::: "memory");
}

// Feature-map element d (0..287) of a 16-wide raw vector stored as bf16 in LDS.
__device__ __forceinline__ float featval(const __bf16* xrow, int d) {
  if (d == 0) return 1.0f;
  if (d < 17) return (float)xrow[d - 1] * C_LIN;
  if (d < 273) {
    int t = d - 17;
    return (float)xrow[t >> 4] * (float)xrow[t & 15] * C_QUAD;
  }
  return 0.0f;  // padding lanes
}

// Two 8-wide K-run fragment load per the CDNA5 16-bit A/B VGPR layout:
// lanes 0-15 hold K [kh*8 .. kh*8+8) and K [16+kh*8 .. +8) per half-wave.
__device__ __forceinline__ v16bf load_frag(const __bf16* p0, const __bf16* p1) {
  v16bf v;
#pragma unroll
  for (int i = 0; i < 8; ++i) { v[i] = p0[i]; v[8 + i] = p1[i]; }
  return v;
}

// Pack two stacked 16x16 fp32 C tiles into one 32x16 bf16 B fragment
// (per-lane repack only; C layout matches B fragment K-runs exactly).
__device__ __forceinline__ v16bf pack_S_frag(v8f c0, v8f c1) {
  v16bf b;
#pragma unroll
  for (int i = 0; i < 8; ++i) { b[i] = (__bf16)c0[i]; b[8 + i] = (__bf16)c1[i]; }
  return b;
}

// ---------------------------------------------------------------------------
// fp32 -> bf16 streaming conversion (float4 in, 4x bf16 out)
// ---------------------------------------------------------------------------
__global__ __launch_bounds__(256) void cvt_f32_bf16_kernel(
    const float* __restrict__ in, __bf16* __restrict__ out, int n4) {
  int i = blockIdx.x * 256 + threadIdx.x;
  if (i < n4) {
    float4 v = ((const float4*)in)[i];
    v4bf o;
    o[0] = (__bf16)v.x; o[1] = (__bf16)v.y; o[2] = (__bf16)v.z; o[3] = (__bf16)v.w;
    ((v4bf*)out)[i] = o;
  }
}

// ---------------------------------------------------------------------------
// C[M,N] = A[M,K] @ W[N,K]^T   (bf16 in, bf16 or fp32 out)
// 256 threads (8 waves), block tile 256x64, wave tile 32x64, K-step 32.
// Async global->LDS staging, double buffered.
// ---------------------------------------------------------------------------
template <bool OUT_BF16>
__global__ __launch_bounds__(256) void wmma_gemm_bf16(
    const __bf16* __restrict__ A, const __bf16* __restrict__ W,
    void* __restrict__ Cout, int M, int N, int K) {
  __shared__ __align__(16) __bf16 As[2][256][40];  // +8 pad, rows 80B (16B mult)
  __shared__ __align__(16) __bf16 Ws[2][64][40];

  const int tid  = threadIdx.x;
  const int w    = tid >> 5;
  const int lane = tid & 31;
  const int lr   = lane & 15;
  const int kh   = lane >> 4;
  const int m0   = blockIdx.x * 256;
  const int n0   = blockIdx.y * 64;

  v8f acc[2][4];
#pragma unroll
  for (int dt = 0; dt < 2; ++dt)
#pragma unroll
    for (int nt = 0; nt < 4; ++nt) acc[dt][nt] = {};

  auto stage = [&](int buf, int k0) {
    // A tile: 256 rows x 32 cols bf16 = 1024 chunks of 16B
#pragma unroll
    for (int s = 0; s < 4; ++s) {
      int idx = tid + s * 256;
      int r = idx >> 2, c4 = idx & 3;
      async_ld16(&As[buf][r][c4 * 8], A + (size_t)(m0 + r) * K + k0 + c4 * 8);
    }
    // W tile: 64 rows x 32 cols = 256 chunks of 16B
    {
      int r = tid >> 2, c4 = tid & 3;
      async_ld16(&Ws[buf][r][c4 * 8], W + (size_t)(n0 + r) * K + k0 + c4 * 8);
    }
  };

  stage(0, 0);
  wait_async0();
  __syncthreads();

  const int nk = K >> 5;
  for (int ks = 0; ks < nk; ++ks) {
    const int buf = ks & 1;
    if (ks + 1 < nk) stage(buf ^ 1, (ks + 1) * 32);

    v16bf bfr[4];
#pragma unroll
    for (int nt = 0; nt < 4; ++nt) {
      const __bf16* br = &Ws[buf][nt * 16 + lr][0];
      bfr[nt] = load_frag(&br[kh * 8], &br[16 + kh * 8]);
    }
#pragma unroll
    for (int dt = 0; dt < 2; ++dt) {
      const __bf16* ar = &As[buf][w * 32 + dt * 16 + lr][0];
      v16bf a = load_frag(&ar[kh * 8], &ar[16 + kh * 8]);
#pragma unroll
      for (int nt = 0; nt < 4; ++nt) acc[dt][nt] = wmma_bf16(a, bfr[nt], acc[dt][nt]);
    }
    wait_async0();
    __syncthreads();
  }

#pragma unroll
  for (int dt = 0; dt < 2; ++dt)
#pragma unroll
    for (int nt = 0; nt < 4; ++nt)
#pragma unroll
      for (int r = 0; r < 8; ++r) {
        size_t off = (size_t)(m0 + w * 32 + dt * 16 + r + 8 * kh) * N + n0 + nt * 16 + lr;
        if (OUT_BF16) ((__bf16*)Cout)[off] = (__bf16)acc[dt][nt][r];
        else          ((float*)Cout)[off]  = acc[dt][nt][r];
      }
}

// ---------------------------------------------------------------------------
// Chunked causal Taylor linear attention. One block per (batch, head).
// 288 threads = 9 waves. Running state S (288x64 fp32) lives as WMMA
// accumulators: wave w owns d-rows [32w, 32w+32) => 2x4 v8f tiles.
// ---------------------------------------------------------------------------
__global__ __launch_bounds__(288) void taylor_attn_kernel(
    const __bf16* __restrict__ Q, const __bf16* __restrict__ Kp,
    const __bf16* __restrict__ V, __bf16* __restrict__ Y) {
  const int bh   = blockIdx.x;
  const int b    = bh >> 4;
  const int h    = bh & 15;
  const int tid  = threadIdx.x;
  const int w    = tid >> 5;
  const int lane = tid & 31;
  const int lr   = lane & 15;
  const int kh   = lane >> 4;

  __shared__ __align__(16) __bf16 q_s[CHUNKL][16];      // rows 32B
  __shared__ __align__(16) __bf16 k_s[CHUNKL][16];
  __shared__ __align__(16) __bf16 v_t[HDIM][136];       // v transposed, rows 272B
  __shared__ __align__(16) __bf16 attn_s[CHUNKL][136];
  __shared__ float num_s[CHUNKL][HDIM];
  __shared__ float den_s[CHUNKL];
  __shared__ float z_s[DE_PAD];

  const __bf16* qg = Q + (size_t)b * SEQ_L * (NHEADS * NFEAT) + h * NFEAT;
  const __bf16* kg = Kp + (size_t)b * SEQ_L * (NHEADS * NFEAT) + h * NFEAT;
  const __bf16* vg = V + (size_t)b * SEQ_L * (NHEADS * HDIM) + h * HDIM;
  __bf16*       yg = Y + (size_t)b * SEQ_L * (NHEADS * HDIM) + h * HDIM;

  v8f S[2][4];
#pragma unroll
  for (int dt = 0; dt < 2; ++dt)
#pragma unroll
    for (int et = 0; et < 4; ++et) S[dt][et] = {};
  z_s[tid] = 0.0f;

  for (int c = 0; c < NCHUNK; ++c) {
    const int l0 = c * CHUNKL;

    // ---- phase 0: stage chunk tiles (q/k via async b128), zero numerator ----
    for (int idx = tid; idx < 512; idx += 288) {
      int r = (idx & 255) >> 1, cc = idx & 1;
      if (idx < 256) async_ld16(&q_s[r][cc * 8], qg + (size_t)(l0 + r) * (NHEADS * NFEAT) + cc * 8);
      else           async_ld16(&k_s[r][cc * 8], kg + (size_t)(l0 + r) * (NHEADS * NFEAT) + cc * 8);
    }
    for (int idx = tid; idx < CHUNKL * HDIM; idx += 288) {
      int r = idx >> 6, e = idx & 63;
      v_t[e][r]   = vg[(size_t)(l0 + r) * (NHEADS * HDIM) + e];
      num_s[r][e] = 0.0f;
    }
    wait_async0();
    __syncthreads();

    // ---- phase 1: attn = mask * (1 + t/4 + t^2/32), t = q.k (K=16 pad 32) ----
    if (w < 8) {
      const int m = w * 16;
      const __bf16* qrow = &q_s[m + lr][0];
      v16bf a;
#pragma unroll
      for (int i = 0; i < 8; ++i) { a[i] = qrow[kh * 8 + i]; a[8 + i] = (__bf16)0.0f; }
      for (int nt = 0; nt < 8; ++nt) {
        const __bf16* krow = &k_s[nt * 16 + lr][0];
        v16bf bb;
#pragma unroll
        for (int i = 0; i < 8; ++i) { bb[i] = krow[kh * 8 + i]; bb[8 + i] = (__bf16)0.0f; }
        v8f t = {};
        t = wmma_bf16(a, bb, t);
#pragma unroll
        for (int r = 0; r < 8; ++r) {
          int rg = m + r + 8 * kh;
          int cg = nt * 16 + lr;
          float tv  = t[r];
          float val = (rg >= cg) ? (1.0f + 0.25f * tv + tv * tv * 0.03125f) : 0.0f;
          attn_s[rg][cg] = (__bf16)val;
        }
      }
    }
    __syncthreads();

    // ---- phase 1b: den = rowsum(attn) + qf . z   (OLD z) ----
    if (tid < CHUNKL) {
      float qv[16];
#pragma unroll
      for (int f = 0; f < 16; ++f) qv[f] = (float)q_s[tid][f];
      float s = 0.0f;
      for (int j = 0; j < CHUNKL; ++j) s += (float)attn_s[tid][j];
      float dz = z_s[0];
#pragma unroll
      for (int ai = 0; ai < 16; ++ai) dz += z_s[1 + ai] * qv[ai] * C_LIN;
      for (int ai = 0; ai < 16; ++ai) {
        float qa = qv[ai] * C_QUAD;
#pragma unroll
        for (int bi = 0; bi < 16; ++bi) dz += z_s[17 + ai * 16 + bi] * qa * qv[bi];
      }
      den_s[tid] = s + dz;
    }

    // ---- phase 2: inter numerator qf @ S_old (all 9 waves, LDS atomics) ----
    {
      const int dbase = w * 32;
      v16bf bS[4];
#pragma unroll
      for (int et = 0; et < 4; ++et) bS[et] = pack_S_frag(S[0][et], S[1][et]);
      for (int mt = 0; mt < 8; ++mt) {
        const __bf16* qrow = &q_s[mt * 16 + lr][0];
        v16bf a;
#pragma unroll
        for (int i = 0; i < 8; ++i) {
          a[i]     = (__bf16)featval(qrow, dbase + kh * 8 + i);
          a[8 + i] = (__bf16)featval(qrow, dbase + 16 + kh * 8 + i);
        }
#pragma unroll
        for (int et = 0; et < 4; ++et) {
          v8f t = {};
          t = wmma_bf16(a, bS[et], t);
#pragma unroll
          for (int r = 0; r < 8; ++r)
            atomicAdd(&num_s[mt * 16 + r + 8 * kh][et * 16 + lr], t[r]);
        }
      }
    }

    // ---- phase 3: intra numerator attn @ v ----
    if (w < 8) {
      const int m = w * 16;
#pragma unroll
      for (int et = 0; et < 4; ++et) {
        v8f t = {};
        for (int ks = 0; ks < 4; ++ks) {
          const __bf16* arow = &attn_s[m + lr][ks * 32];
          const __bf16* vrow = &v_t[et * 16 + lr][ks * 32];
          v16bf a  = load_frag(&arow[kh * 8], &arow[16 + kh * 8]);
          v16bf bb = load_frag(&vrow[kh * 8], &vrow[16 + kh * 8]);
          t = wmma_bf16(a, bb, t);
        }
#pragma unroll
        for (int r = 0; r < 8; ++r)
          atomicAdd(&num_s[m + r + 8 * kh][et * 16 + lr], t[r]);
      }
    }
    __syncthreads();

    // ---- phase 4: y = num / (den + eps)  (bf16 out) ----
    for (int idx = tid; idx < CHUNKL * HDIM; idx += 288) {
      int r = idx >> 6, e = idx & 63;
      yg[(size_t)(l0 + r) * (NHEADS * HDIM) + e] =
          (__bf16)(num_s[r][e] / (den_s[r] + 1e-12f));
    }

    // ---- phase 5a: S += kf^T @ v  (register accumulators) ----
    for (int ks = 0; ks < 4; ++ks) {
      v16bf bb[4];
#pragma unroll
      for (int et = 0; et < 4; ++et) {
        const __bf16* vrow = &v_t[et * 16 + lr][ks * 32];
        bb[et] = load_frag(&vrow[kh * 8], &vrow[16 + kh * 8]);
      }
#pragma unroll
      for (int dt = 0; dt < 2; ++dt) {
        const int d = w * 32 + dt * 16 + lr;
        v16bf a;
#pragma unroll
        for (int i = 0; i < 8; ++i) {
          int j1 = ks * 32 + kh * 8 + i;
          int j2 = ks * 32 + 16 + kh * 8 + i;
          a[i]     = (__bf16)featval(&k_s[j1][0], d);
          a[8 + i] = (__bf16)featval(&k_s[j2][0], d);
        }
#pragma unroll
        for (int et = 0; et < 4; ++et) S[dt][et] = wmma_bf16(a, bb[et], S[dt][et]);
      }
    }

    // ---- phase 5b: z += colsum(kf) ----
    if (tid < 273) {
      float s = 0.0f;
      for (int j = 0; j < CHUNKL; ++j) s += featval(&k_s[j][0], tid);
      z_s[tid] += s;
    }
    __syncthreads();
  }
}

// ---------------------------------------------------------------------------
extern "C" void kernel_launch(void* const* d_in, const int* in_sizes, int n_in,
                              void* d_out, int out_size, void* d_ws, size_t ws_size,
                              hipStream_t stream) {
  (void)in_sizes; (void)n_in; (void)out_size; (void)ws_size;
  const float* hidden = (const float*)d_in[0];
  const float* Wq     = (const float*)d_in[1];
  const float* Wk     = (const float*)d_in[2];
  const float* Wv     = (const float*)d_in[3];
  const float* Wo     = (const float*)d_in[4];
  float*       out    = (float*)d_out;

  const int M = NBATCH * SEQ_L;  // 8192

  // bf16 workspace layout
  __bf16* hid_bf = (__bf16*)d_ws;                    // M x 1024
  __bf16* wq_bf  = hid_bf + (size_t)M * 1024;        // 256 x 1024
  __bf16* wk_bf  = wq_bf + (size_t)256 * 1024;
  __bf16* wv_bf  = wk_bf + (size_t)256 * 1024;       // 1024 x 1024
  __bf16* wo_bf  = wv_bf + (size_t)1024 * 1024;
  __bf16* q_bf   = wo_bf + (size_t)1024 * 1024;      // M x 256
  __bf16* k_bf   = q_bf + (size_t)M * 256;
  __bf16* v_bf   = k_bf + (size_t)M * 256;           // M x 1024
  __bf16* y_bf   = v_bf + (size_t)M * 1024;          // M x 1024

  auto cvt = [&](const float* src, __bf16* dst, int n) {
    int n4 = n >> 2;
    cvt_f32_bf16_kernel<<<(n4 + 255) / 256, 256, 0, stream>>>(src, dst, n4);
  };
  cvt(hidden, hid_bf, M * 1024);
  cvt(Wq, wq_bf, 256 * 1024);
  cvt(Wk, wk_bf, 256 * 1024);
  cvt(Wv, wv_bf, 1024 * 1024);
  cvt(Wo, wo_bf, 1024 * 1024);

  dim3 blk(256);
  wmma_gemm_bf16<true><<<dim3(M / 256, 256 / 64), blk, 0, stream>>>(hid_bf, wq_bf, q_bf, M, 256, 1024);
  wmma_gemm_bf16<true><<<dim3(M / 256, 256 / 64), blk, 0, stream>>>(hid_bf, wk_bf, k_bf, M, 256, 1024);
  wmma_gemm_bf16<true><<<dim3(M / 256, 1024 / 64), blk, 0, stream>>>(hid_bf, wv_bf, v_bf, M, 1024, 1024);

  taylor_attn_kernel<<<dim3(NBATCH * NHEADS), dim3(288), 0, stream>>>(q_bf, k_bf, v_bf, y_bf);

  wmma_gemm_bf16<false><<<dim3(M / 256, 1024 / 64), blk, 0, stream>>>(y_bf, wo_bf, out, M, 1024, 1024);
}
